// FasterSPFNSMultiHeadAttention_89575837925543
// MI455X (gfx1250) — matrix operand order repaired and checked by reading, not compiled, not run on hardware
//
#include <hip/hip_runtime.h>
#include <math.h>

// ---------------------------------------------------------------------------
// Problem constants (from reference)
// ---------------------------------------------------------------------------
static constexpr int Bb   = 2;
static constexpr int S    = 2048;
static constexpr int H    = 1024;
static constexpr int NH   = 16;
static constexpr int HD   = 64;
static constexpr int M    = Bb * S;          // 4096 token rows
static constexpr int NQKV = 3 * H;           // 3072

typedef __attribute__((ext_vector_type(16))) __bf16 bf16x16;
typedef __attribute__((ext_vector_type(8)))  float  f32x8;

#define WMMA_BF16(a, b, c) \
  __builtin_amdgcn_wmma_f32_16x16x32_bf16(false, (a), false, (b), (short)0, (c), false, false)

union FragU { bf16x16 v; uint4 u[2]; };

// A fragment (16x32 bf16, M=lane&15, K split 0..7/16..23 vs 8..15/24..31)
__device__ __forceinline__ bf16x16 load_a(const __bf16* tile, int ld) {
  int lane = threadIdx.x & 31;
  int row  = lane & 15;
  int ko   = (lane >> 4) << 3;                  // 0 or 8
  const __bf16* p = tile + (size_t)row * ld + ko;
  FragU f;
  f.u[0] = *reinterpret_cast<const uint4*>(p);
  f.u[1] = *reinterpret_cast<const uint4*>(p + 16);
  return f.v;
}

// B fragment (32x16 bf16, N=lane&15, K split 0..7/16..23 vs 8..15/24..31)
__device__ __forceinline__ bf16x16 load_b(const __bf16* tile, int ld) {
  int lane = threadIdx.x & 31;
  int row  = lane & 15;                         // N
  int ko   = (lane >> 4) << 3;                  // 0 or 8
  const __bf16* p = tile + (size_t)row * ld + ko;
  FragU f;
  f.u[0] = *reinterpret_cast<const uint4*>(p);
  f.u[1] = *reinterpret_cast<const uint4*>(p + 16);
  return f.v;
}

// Fast acos (Hastings): max abs error ~6.8e-5 rad -- far below bf16 operand
// noise.  3 FMA + v_sqrt + select instead of the ~30-op ocml polynomial.
__device__ __forceinline__ float fast_acos(float x) {
  float a = fabsf(x);
  float p = -0.0187293f;
  p = p * a + 0.0742610f;
  p = p * a - 0.2121144f;
  p = p * a + 1.5707288f;
  float r = p * __builtin_amdgcn_sqrtf(1.0f - a);
  return (x < 0.0f) ? (3.14159265358979f - r) : r;
}

// attn_score = (1 + arccos(clip(cos)))^(-3.2)   (BANDWIDTH=1, R=1, -D-ALPHA=-3.2)
// pow via single v_log_f32 / v_exp_f32 (base-2): exp2(-3.2 * log2(1+g)).
__device__ __forceinline__ float attn_fn(float c) {
  c = fminf(fmaxf(c, -1.0f + 1e-7f), 1.0f - 1e-7f);
  float g = fast_acos(c);
  return __builtin_amdgcn_exp2f(-3.2f * __builtin_amdgcn_logf(1.0f + g));
}

// ---------------------------------------------------------------------------
// fp32 -> bf16 convert
// ---------------------------------------------------------------------------
__global__ void cvt_bf16(const float* __restrict__ src, __bf16* __restrict__ dst, int n) {
  int i = blockIdx.x * 256 + threadIdx.x;
  if (i < n) dst[i] = (__bf16)src[i];
}

// ---------------------------------------------------------------------------
// C[M,N] = A[M,K] * Bw[N,K]^T + bias[N]   (bf16 WMMA, f32 accumulate)
// Block: 128 threads = 4 waves; wave w does rows [by*64+w*16, +16), cols [bx*64, +64)
// ---------------------------------------------------------------------------
template <bool OUT_BF16>
__global__ void gemm_wmma(const __bf16* __restrict__ A, const __bf16* __restrict__ Bw,
                          const float* __restrict__ bias, void* __restrict__ Cout,
                          int Mm, int Nn, int Kk) {
  int w = threadIdx.x >> 5, lane = threadIdx.x & 31;
  int n0 = blockIdx.x * 64;
  int s0 = blockIdx.y * 64 + w * 16;
  f32x8 acc[4];
#pragma unroll
  for (int j = 0; j < 4; ++j) acc[j] = f32x8{0, 0, 0, 0, 0, 0, 0, 0};

  for (int k = 0; k < Kk; k += 32) {
    bf16x16 a = load_a(A + (size_t)s0 * Kk + k, Kk);
#pragma unroll
    for (int j = 0; j < 4; ++j) {
      bf16x16 b = load_b(Bw + (size_t)(n0 + j * 16) * Kk + k, Kk);
      acc[j] = WMMA_BF16(a, b, acc[j]);
    }
  }

  int hi8 = (lane >> 4) << 3, nl = lane & 15;
#pragma unroll
  for (int j = 0; j < 4; ++j) {
    int n = n0 + j * 16 + nl;
    float bb = bias[n];
#pragma unroll
    for (int v = 0; v < 8; ++v) {
      int mrow = s0 + v + hi8;
      float val = acc[j][v] + bb;
      if constexpr (OUT_BF16)
        ((__bf16*)Cout)[(size_t)mrow * Nn + n] = (__bf16)val;
      else
        ((float*)Cout)[(size_t)mrow * Nn + n] = val;
    }
  }
}

// ---------------------------------------------------------------------------
// Split qkv, L2-normalize q/k per (b,s,h) 64-vector, store q/k as [bh,s,64],
// v transposed as vT[bh,d,S]. One wave per (b,s,h); lane handles d, d+32.
// ---------------------------------------------------------------------------
__global__ void norm_split(const __bf16* __restrict__ qkv, __bf16* __restrict__ qn,
                           __bf16* __restrict__ kn, __bf16* __restrict__ vT) {
  int w = threadIdx.x >> 5, lane = threadIdx.x & 31;
  int idx = blockIdx.x * 4 + w;        // (b*S + s)*16 + h
  int h  = idx & 15;
  int bs = idx >> 4;
  int b = bs >> 11, s = bs & (S - 1);
  int bh = b * NH + h;

  const __bf16* row = qkv + (size_t)bs * NQKV + h * HD;
  float q0 = (float)row[lane],        q1 = (float)row[lane + 32];
  float k0 = (float)row[H + lane],    k1 = (float)row[H + lane + 32];
  float v0 = (float)row[2*H + lane],  v1 = (float)row[2*H + lane + 32];

  float sq = q0 * q0 + q1 * q1;
  float sk = k0 * k0 + k1 * k1;
#pragma unroll
  for (int off = 16; off; off >>= 1) {
    sq += __shfl_xor(sq, off, 32);
    sk += __shfl_xor(sk, off, 32);
  }
  float rq = 1.0f / fmaxf(sqrtf(sq), 1e-12f);
  float rk = 1.0f / fmaxf(sqrtf(sk), 1e-12f);

  size_t base = ((size_t)bh * S + s) * HD;
  qn[base + lane]      = (__bf16)(q0 * rq);
  qn[base + lane + 32] = (__bf16)(q1 * rq);
  kn[base + lane]      = (__bf16)(k0 * rk);
  kn[base + lane + 32] = (__bf16)(k1 * rk);
  vT[((size_t)bh * HD + lane)      * S + s] = (__bf16)v0;
  vT[((size_t)bh * HD + lane + 32) * S + s] = (__bf16)v1;
}

// ---------------------------------------------------------------------------
// Pass A: rncol[bh,t] = 1 / sum_s attn(q_s . k_t).  Block = (bh, t-tile of 16);
// 4 waves split the s range; WMMA for q.kT; LDS reduce.
// ---------------------------------------------------------------------------
__global__ void attn_colsum(const __bf16* __restrict__ qn, const __bf16* __restrict__ kn,
                            float* __restrict__ rncol) {
  int w = threadIdx.x >> 5, lane = threadIdx.x & 31;
  int bh = blockIdx.x;
  int t0 = blockIdx.y * 16;

  const __bf16* kt = kn + ((size_t)bh * S + t0) * HD;
  bf16x16 bk0 = load_b(kt, HD);        // d = 0..31
  bf16x16 bk1 = load_b(kt + 32, HD);   // d = 32..63

  float colp = 0.0f;
  for (int st = 0; st < 32; ++st) {
    int s0 = w * 512 + st * 16;
    const __bf16* qs = qn + ((size_t)bh * S + s0) * HD;
    bf16x16 a0 = load_a(qs, HD);
    bf16x16 a1 = load_a(qs + 32, HD);
    f32x8 c = f32x8{0, 0, 0, 0, 0, 0, 0, 0};
    c = WMMA_BF16(a0, bk0, c);
    c = WMMA_BF16(a1, bk1, c);
#pragma unroll
    for (int v = 0; v < 8; ++v) colp += attn_fn(c[v]);
  }
  colp += __shfl_xor(colp, 16, 32);    // combine the two M-halves of each column

  __shared__ float cs[4][16];
  if (lane < 16) cs[w][lane] = colp;
  __syncthreads();
  if (threadIdx.x < 16) {
    int t = threadIdx.x;
    float sum = cs[0][t] + cs[1][t] + cs[2][t] + cs[3][t];
    rncol[(size_t)bh * S + t0 + t] = 1.0f / sum;   // store n_c^-1 (A = 1)
  }
}

// ---------------------------------------------------------------------------
// Pass B: out[s,:] = (sum_t attn*nc^-1 * v_t) / (sum_t attn*nc^-1).
// Block = (bh, s-tile of 16); 4 waves split t; recompute score tiles with WMMA,
// reshape P through per-wave LDS tile, P.v and row-sum (ones-fragment) via WMMA.
// ---------------------------------------------------------------------------
__global__ void attn_out(const __bf16* __restrict__ qn, const __bf16* __restrict__ kn,
                         const __bf16* __restrict__ vT, const float* __restrict__ rncol,
                         __bf16* __restrict__ ao) {
  __shared__ alignas(16) __bf16 pt[4][16][32];
  __shared__ float redV[4][16][64];
  __shared__ float redR[4][16];

  int w = threadIdx.x >> 5, lane = threadIdx.x & 31;
  int bh = blockIdx.x, b = bh >> 4, h = bh & 15;
  int s0 = blockIdx.y * 16;
  int hi8 = (lane >> 4) << 3, nl = lane & 15;

  const __bf16* qs = qn + ((size_t)bh * S + s0) * HD;
  bf16x16 aq0 = load_a(qs, HD);
  bf16x16 aq1 = load_a(qs + 32, HD);

  bf16x16 ones;
#pragma unroll
  for (int e = 0; e < 16; ++e) ones[e] = (__bf16)1.0f;

  f32x8 accV[4];
#pragma unroll
  for (int j = 0; j < 4; ++j) accV[j] = f32x8{0, 0, 0, 0, 0, 0, 0, 0};
  f32x8 accR = f32x8{0, 0, 0, 0, 0, 0, 0, 0};

  for (int tc = 0; tc < 16; ++tc) {
    int t0 = w * 512 + tc * 32;
    // two 16x16 score sub-tiles -> P = attn * nc^-1, staged to LDS as [16s][32t]
#pragma unroll
    for (int half = 0; half < 2; ++half) {
      const __bf16* kt = kn + ((size_t)bh * S + t0 + half * 16) * HD;
      f32x8 c = f32x8{0, 0, 0, 0, 0, 0, 0, 0};
      c = WMMA_BF16(aq0, load_b(kt, HD), c);
      c = WMMA_BF16(aq1, load_b(kt + 32, HD), c);
      float inv = rncol[(size_t)bh * S + t0 + half * 16 + nl];  // n_c^-1 (A = 1)
#pragma unroll
      for (int v = 0; v < 8; ++v) {
        float p = attn_fn(c[v]) * inv;
        pt[w][v + hi8][half * 16 + nl] = (__bf16)p;
      }
    }
    asm volatile("s_wait_dscnt 0" ::: "memory");   // LDS writes visible to whole wave

    bf16x16 pA = load_a(&pt[w][0][0], 32);         // P as A-operand (K = t-chunk)
#pragma unroll
    for (int j = 0; j < 4; ++j) {
      bf16x16 bv = load_b(vT + ((size_t)bh * HD + j * 16) * S + t0, S);
      accV[j] = WMMA_BF16(pA, bv, accV[j]);
    }
    accR = WMMA_BF16(pA, ones, accR);              // row-sum of P
    asm volatile("s_wait_dscnt 0" ::: "memory");   // reads done before next overwrite
  }

  // cross-wave reduction + row normalization
#pragma unroll
  for (int j = 0; j < 4; ++j)
#pragma unroll
    for (int v = 0; v < 8; ++v) redV[w][v + hi8][j * 16 + nl] = accV[j][v];
  if (nl == 0) {
#pragma unroll
    for (int v = 0; v < 8; ++v) redR[w][v + hi8] = accR[v];
  }
  __syncthreads();

  for (int e = threadIdx.x; e < 16 * 64; e += 128) {
    int sl = e >> 6, d = e & 63;
    float sum = redV[0][sl][d] + redV[1][sl][d] + redV[2][sl][d] + redV[3][sl][d];
    float rs  = redR[0][sl] + redR[1][sl] + redR[2][sl] + redR[3][sl];
    ao[(size_t)(b * S + s0 + sl) * H + h * HD + d] = (__bf16)(sum / fmaxf(rs, 1e-12f));
  }
}

// ---------------------------------------------------------------------------
// Host launcher
// ---------------------------------------------------------------------------
extern "C" void kernel_launch(void* const* d_in, const int* in_sizes, int n_in,
                              void* d_out, int out_size, void* d_ws, size_t ws_size,
                              hipStream_t stream) {
  const float* x     = (const float*)d_in[0];
  const float* w_qkv = (const float*)d_in[1];
  const float* b_qkv = (const float*)d_in[2];
  const float* w_out = (const float*)d_in[3];
  const float* b_out = (const float*)d_in[4];
  float* out = (float*)d_out;

  char* ws = (char*)d_ws;
  size_t off = 0;
  auto alloc = [&](size_t bytes) -> void* {
    void* p = ws + off;
    off += (bytes + 255) & ~(size_t)255;
    return p;
  };
  __bf16* xb    = (__bf16*)alloc((size_t)M * H * 2);        //  8.4 MB
  __bf16* wqkvb = (__bf16*)alloc((size_t)NQKV * H * 2);     //  6.3 MB
  __bf16* woutb = (__bf16*)alloc((size_t)H * H * 2);        //  2.1 MB
  __bf16* qkv   = (__bf16*)alloc((size_t)M * NQKV * 2);     // 25.2 MB
  __bf16* qn    = (__bf16*)alloc((size_t)M * H * 2);        //  8.4 MB
  __bf16* knb   = (__bf16*)alloc((size_t)M * H * 2);        //  8.4 MB
  __bf16* vT    = (__bf16*)alloc((size_t)M * H * 2);        //  8.4 MB
  float*  rncol = (float*)alloc((size_t)Bb * NH * S * 4);   //  0.3 MB
  __bf16* ao    = (__bf16*)alloc((size_t)M * H * 2);        //  8.4 MB
  (void)ws_size; (void)in_sizes; (void)n_in; (void)out_size;

  cvt_bf16<<<(M * H + 255) / 256, 256, 0, stream>>>(x, xb, M * H);
  cvt_bf16<<<(NQKV * H + 255) / 256, 256, 0, stream>>>(w_qkv, wqkvb, NQKV * H);
  cvt_bf16<<<(H * H + 255) / 256, 256, 0, stream>>>(w_out, woutb, H * H);

  gemm_wmma<true><<<dim3(NQKV / 64, M / 64), 128, 0, stream>>>(xb, wqkvb, b_qkv, qkv, M, NQKV, H);
  norm_split<<<(M * NH) / 4, 128, 0, stream>>>(qkv, qn, knb, vT);
  attn_colsum<<<dim3(Bb * NH, S / 16), 128, 0, stream>>>(qn, knb, rncol);
  attn_out<<<dim3(Bb * NH, S / 16), 128, 0, stream>>>(qn, knb, vT, rncol, ao);
  gemm_wmma<false><<<dim3(H / 64, M / 64), 128, 0, stream>>>(ao, woutb, b_out, out, M, H, H);
}
